// Block_6373731467375
// MI455X (gfx1250) — matrix operand order repaired and checked by reading, not compiled, not run on hardware
//
#include <hip/hip_runtime.h>

// ---------------------------------------------------------------------------
// Types for CDNA5 WMMA (wave32, 16x16x32 f16 -> f32 accumulate)
// ---------------------------------------------------------------------------
typedef __attribute__((ext_vector_type(8)))  _Float16 h8;
typedef __attribute__((ext_vector_type(4)))  _Float16 h4;
typedef __attribute__((ext_vector_type(16))) _Float16 h16v;
typedef __attribute__((ext_vector_type(8)))  float    f8;
typedef __attribute__((ext_vector_type(4)))  float    f4;

__device__ inline h16v cat8(h8 lo, h8 hi) {
    return __builtin_shufflevector(lo, hi, 0,1,2,3,4,5,6,7,8,9,10,11,12,13,14,15);
}

// A-matrix 16x32 f16 fragment from LDS (row-major, stride in halfs).
// ISA layout: lanes 0-15 rows M=0..15 hold K chunks {0..7,16..23};
//             lanes 16-31 same rows hold K chunks {8..15,24..31}.
__device__ inline h16v loadA(const _Float16* p, int stride, int row0, int k0) {
    const int lane = threadIdx.x & 31;
    const _Float16* q = p + (size_t)(row0 + (lane & 15)) * stride + k0 + ((lane >> 4) << 3);
    h8 lo = *(const h8*)q;
    h8 hi = *(const h8*)(q + 16);
    return cat8(lo, hi);
}

// B-matrix 32x16 f16 fragment from LDS stored as Bt[N][K] (row-major, stride halfs).
// ISA layout: lanes 0-15 col N=0..15 hold K chunks {0..7,16..23}; lanes 16-31 hold K chunks {8..15,24..31}.
__device__ inline h16v loadB(const _Float16* p, int stride, int col0, int k0) {
    const int lane = threadIdx.x & 31;
    const _Float16* q = p + (size_t)(col0 + (lane & 15)) * stride + k0 + ((lane >> 4) << 3);
    h8 lo = *(const h8*)q;
    h8 hi = *(const h8*)(q + 16);
    return cat8(lo, hi);
}

__device__ inline f8 wmma32(h16v a, h16v b, f8 c) {
    return __builtin_amdgcn_wmma_f32_16x16x32_f16(false, a, false, b, (short)0, c, false, false);
}

// ---------------------------------------------------------------------------
// One-time weight conversion: f32 [K,N] row-major -> f16 transposed [N,K].
// Lets the GEMM stage B with b128 loads and no per-tile conversion.
// ---------------------------------------------------------------------------
__global__ __launch_bounds__(256)
void transpose_w16(const float* __restrict__ w, _Float16* __restrict__ out, int K, int N)
{
    const int idx = blockIdx.x * 256 + threadIdx.x;
    if (idx >= K * N) return;
    const int k = idx / N, n = idx - k * N;
    out[(size_t)n * K + k] = (_Float16)w[idx];
}

// ---------------------------------------------------------------------------
// LayerNorm -> f16, vectorized (float4 in, h4 out).
// ROLLED: output row (b,p) takes src j=(p+64)&2047 of the padded sequence;
// rows with j>=2000 are zeroed (pad tokens). One wave per row.
// ---------------------------------------------------------------------------
template<bool ROLLED>
__global__ __launch_bounds__(256)
void layernorm_f16(const float* __restrict__ in, const float* __restrict__ w,
                   const float* __restrict__ b, _Float16* __restrict__ out, int nrows)
{
    const int lane = threadIdx.x & 31, wid = threadIdx.x >> 5;
    const int row = blockIdx.x * 8 + wid;
    if (row >= nrows) return;

    const float* src = nullptr;
    bool valid = true;
    if (ROLLED) {
        const int bb = row >> 11, p = row & 2047;
        const int j = (p + 64) & 2047;
        valid = (j < 2000);
        if (valid) src = in + (size_t)(bb * 2000 + j) * 768;
    } else {
        src = in + (size_t)row * 768;
    }
    _Float16* dst = out + (size_t)row * 768;
    if (!valid) {
        h4 z = {};
#pragma unroll
        for (int i = 0; i < 6; ++i) *(h4*)(dst + (lane + 32 * i) * 4) = z;
        return;
    }
    f4 r[6];
    float s = 0.f;
#pragma unroll
    for (int i = 0; i < 6; ++i) {
        r[i] = *(const f4*)(src + (lane + 32 * i) * 4);
#pragma unroll
        for (int j = 0; j < 4; ++j) s += r[i][j];
    }
#pragma unroll
    for (int m = 16; m; m >>= 1) s += __shfl_xor(s, m, 32);
    const float mean = s * (1.0f / 768.0f);
    float vq = 0.f;
#pragma unroll
    for (int i = 0; i < 6; ++i)
#pragma unroll
        for (int j = 0; j < 4; ++j) { const float d = r[i][j] - mean; vq += d * d; }
#pragma unroll
    for (int m = 16; m; m >>= 1) vq += __shfl_xor(vq, m, 32);
    const float rstd = rsqrtf(vq * (1.0f / 768.0f) + 1e-5f);
#pragma unroll
    for (int i = 0; i < 6; ++i) {
        const int g = (lane + 32 * i) * 4;
        f4 wv = *(const f4*)(w + g);
        f4 bv = *(const f4*)(b + g);
        h4 o;
#pragma unroll
        for (int j = 0; j < 4; ++j) o[j] = (_Float16)((r[i][j] - mean) * rstd * wv[j] + bv[j]);
        *(h4*)(dst + g) = o;
    }
}

// ---------------------------------------------------------------------------
// WMMA GEMM: C[128x128 tile] = act_f16[M,K] @ wt_f16t[N,K]^T + bias.
// K-tile 64 -> 16 WMMAs per barrier pair. 8 waves (4 in M x 2 in N),
// each wave owns 32x64 = 2x4 WMMA tiles. b128 staging for both operands,
// global_prefetch for the next K tile.
// MODE 0: qkv  -> f16 out (coalesced via LDS)
// MODE 1: proj -> un-roll, crop, out32 = resid + val*ls  (xres)
// MODE 2: fc1  -> exact GELU -> f16 out (coalesced via LDS)
// MODE 3: fc2  -> out32 = resid + val*ls  (final output)
// ---------------------------------------------------------------------------
template<int MODE>
__global__ __launch_bounds__(256)
void gemm_wmma(const _Float16* __restrict__ act, const _Float16* __restrict__ wt,
               const float* __restrict__ bias, _Float16* __restrict__ out16,
               float* __restrict__ out32, const float* __restrict__ resid,
               const float* __restrict__ lscale, int K, int Nw)
{
    __shared__ _Float16 smem[2 * 128 * 72];   // As | Bs ; epilogue reuses as Cs
    _Float16* As = smem;                      // 128 rows x 64 K, stride 72
    _Float16* Bs = smem + 128 * 72;           // 128 cols x 64 K, stride 72

    const int tid = threadIdx.x, lane = tid & 31, wid = tid >> 5;
    const int tM = blockIdx.x * 128, tN = blockIdx.y * 128;
    const int wM = (wid >> 1) * 32, wN = (wid & 1) * 64;

    f8 acc[2][4] = {};

    for (int kb = 0; kb < K; kb += 64) {
        __syncthreads();
#pragma unroll
        for (int i = 0; i < 4; ++i) {
            const int c = tid + 256 * i;            // 1024 chunks of 8 halfs
            const int r = c >> 3, off = (c & 7) << 3;
            *(h8*)(As + r * 72 + off) = *(const h8*)(act + (size_t)(tM + r) * K + kb + off);
            *(h8*)(Bs + r * 72 + off) = *(const h8*)(wt  + (size_t)(tN + r) * K + kb + off);
        }
        if (kb + 64 < K) {                          // prefetch next K tile
            __builtin_prefetch(act + (size_t)(tM + (tid >> 1)) * K + kb + 64 + (tid & 1) * 32, 0, 1);
            __builtin_prefetch(wt  + (size_t)(tN + (tid >> 1)) * K + kb + 64 + (tid & 1) * 32, 0, 1);
        }
        __syncthreads();
#pragma unroll
        for (int kk = 0; kk < 64; kk += 32) {
            h16v a0 = loadA(As, 72, wM, kk);
            h16v a1 = loadA(As, 72, wM + 16, kk);
#pragma unroll
            for (int ni = 0; ni < 4; ++ni) {
                h16v b = loadB(Bs, 72, wN + 16 * ni, kk);
                acc[0][ni] = wmma32(a0, b, acc[0][ni]);
                acc[1][ni] = wmma32(a1, b, acc[1][ni]);
            }
        }
    }

    // --- epilogue (C/D layout: row = v + 8*(lane>=16), col = lane&15) ---
    if (MODE == 0 || MODE == 2) {
        __syncthreads();                            // staging buffers now dead
        _Float16* Cs = smem;                        // 128 x 128 f16 (32 KB)
#pragma unroll
        for (int mi = 0; mi < 2; ++mi)
#pragma unroll
        for (int ni = 0; ni < 4; ++ni)
#pragma unroll
        for (int v = 0; v < 8; ++v) {
            const int r  = wM + 16 * mi + v + 8 * (lane >> 4);
            const int cc = wN + 16 * ni + (lane & 15);
            float val = acc[mi][ni][v] + bias[tN + cc];
            if (MODE == 2) val = 0.5f * val * (1.0f + erff(val * 0.70710678118f));
            Cs[r * 128 + cc] = (_Float16)val;
        }
        __syncthreads();
#pragma unroll
        for (int i = 0; i < 8; ++i) {               // coalesced b128 stores
            const int c = tid + 256 * i;            // 2048 chunks of 8 halfs
            const int r = c >> 4, off = (c & 15) << 3;
            *(h8*)(out16 + (size_t)(tM + r) * Nw + tN + off) = *(const h8*)(Cs + r * 128 + off);
        }
    } else {
#pragma unroll
        for (int mi = 0; mi < 2; ++mi)
#pragma unroll
        for (int ni = 0; ni < 4; ++ni)
#pragma unroll
        for (int v = 0; v < 8; ++v) {
            const int row = tM + wM + 16 * mi + v + 8 * (lane >> 4);
            const int col = tN + wN + 16 * ni + (lane & 15);
            const float val = acc[mi][ni][v] + bias[col];
            if (MODE == 1) {
                const int bb = row >> 11, p = row & 2047;
                const int j = (p + 64) & 2047;      // roll back by +SHIFT
                if (j < 2000) {                      // crop pad tokens
                    const size_t idx = (size_t)(bb * 2000 + j) * 768 + col;
                    out32[idx] = resid[idx] + val * lscale[col];
                }
            } else {
                const size_t idx = (size_t)row * Nw + col;
                out32[idx] = resid[idx] + val * lscale[col];
            }
        }
    }
}

// ---------------------------------------------------------------------------
// Window attention: one block per (window, head). 256 threads = 8 waves.
// Wave w owns query rows 16w..16w+15 for both S=QK^T and O=P@V, so softmax
// (scale + rel-bias + mask) stays in registers with half-wave shuffles.
// Output routed through LDS for coalesced b128 stores.
// ---------------------------------------------------------------------------
__global__ __launch_bounds__(256)
void window_attn(const _Float16* __restrict__ qkv, const float* __restrict__ rel_bias,
                 _Float16* __restrict__ out)
{
    __shared__ __align__(16) char smem[54272];
    _Float16* Qs  = (_Float16*)smem;             // 128 x 72   (18432 B)
    _Float16* Ks  = (_Float16*)(smem + 18432);   // 128 x 72   (18432 B)
    _Float16* Vst = (_Float16*)(smem + 36864);   // 64  x 136  (17408 B), V^T
    _Float16* Ps  = (_Float16*)smem;             // 128 x 136  (34816 B), aliases Qs+Ks
    _Float16* Os  = (_Float16*)smem;             // 128 x 72   (18432 B), aliases Ps

    const int tid = threadIdx.x, lane = tid & 31, wid = tid >> 5;
    const int w = blockIdx.x / 12, h = blockIdx.x % 12;
    const size_t rowbase = (size_t)w * 128;

    // ---- stage Q, K (row-major) ----
#pragma unroll
    for (int i = 0; i < 2; ++i) {
        const int c = tid + 256 * i;             // 512 chunks of 8 halfs
        const int r = c >> 2, off = (c & 3) << 3;
        h8 q = *(const h8*)(qkv + (rowbase + r) * 2304 + h * 64 + off);
        h8 k = *(const h8*)(qkv + (rowbase + r) * 2304 + 768 + h * 64 + off);
        *(h8*)(Qs + r * 72 + off) = q;
        *(h8*)(Ks + r * 72 + off) = k;
    }
    // ---- stage V transposed: Vst[d][key] ----
#pragma unroll
    for (int i = 0; i < 4; ++i) {
        const int c = tid + 256 * i;             // 1024 chunks of 8 halfs
        const int key = c >> 3, d0 = (c & 7) << 3;
        h8 v = *(const h8*)(qkv + (rowbase + key) * 2304 + 1536 + h * 64 + d0);
#pragma unroll
        for (int j = 0; j < 8; ++j) Vst[(d0 + j) * 136 + key] = v[j];
    }
    __syncthreads();

    // ---- S = Q K^T : wave strip of 16 rows x 128 cols, K-dim 64 ----
    const int q0 = wid * 16;
    f8 acc[8] = {};
    {
        h16v a0 = loadA(Qs, 72, q0, 0);
        h16v a1 = loadA(Qs, 72, q0, 32);
#pragma unroll
        for (int t = 0; t < 8; ++t) {
            h16v b0 = loadB(Ks, 72, 16 * t, 0);
            h16v b1 = loadB(Ks, 72, 16 * t, 32);
            acc[t] = wmma32(a0, b0, acc[t]);
            acc[t] = wmma32(a1, b1, acc[t]);
        }
    }

    // ---- scale + rel-bias + key mask + softmax, in registers ----
    const int wi = w & 15;                        // window index within batch
    const int qrow = q0 + 8 * (lane >> 4);
#pragma unroll
    for (int v = 0; v < 8; ++v) {
        const int q = qrow + v;
        float vals[8];
        float mx = -3.0e38f;
#pragma unroll
        for (int t = 0; t < 8; ++t) {
            const int key = 16 * t + (lane & 15);
            float x = acc[t][v] * 0.125f + rel_bias[(key - q + 127) * 12 + h];
            const int pos = wi * 128 + key;
            const int src = (pos + 64) & 2047;    // rolled -> original position
            if (src >= 2000) x = -1.0e30f;        // masked key
            vals[t] = x;
            mx = fmaxf(mx, x);
        }
#pragma unroll
        for (int m = 1; m < 16; m <<= 1) mx = fmaxf(mx, __shfl_xor(mx, m, 32));
        float sum = 0.f;
#pragma unroll
        for (int t = 0; t < 8; ++t) { vals[t] = __expf(vals[t] - mx); sum += vals[t]; }
#pragma unroll
        for (int m = 1; m < 16; m <<= 1) sum += __shfl_xor(sum, m, 32);
        const float inv = (sum > 0.f) ? (1.0f / sum) : 0.f;   // nan_to_num
#pragma unroll
        for (int t = 0; t < 8; ++t) acc[t][v] = vals[t] * inv;
    }
    __syncthreads();                              // all reads of Qs/Ks done
#pragma unroll
    for (int v = 0; v < 8; ++v)
#pragma unroll
        for (int t = 0; t < 8; ++t)
            Ps[(qrow + v) * 136 + 16 * t + (lane & 15)] = (_Float16)acc[t][v];
    __syncthreads();

    // ---- O = P @ V : 16 rows x 64 cols, K-dim 128 ----
    f8 oac[4] = {};
#pragma unroll
    for (int kk = 0; kk < 128; kk += 32) {
        h16v a = loadA(Ps, 136, q0, kk);
#pragma unroll
        for (int t = 0; t < 4; ++t) {
            h16v b = loadB(Vst, 136, 16 * t, kk);
            oac[t] = wmma32(a, b, oac[t]);
        }
    }
    __syncthreads();                              // Ps reads done
#pragma unroll
    for (int t = 0; t < 4; ++t)
#pragma unroll
        for (int v = 0; v < 8; ++v)
            Os[(qrow + v) * 72 + 16 * t + (lane & 15)] = (_Float16)oac[t][v];
    __syncthreads();
#pragma unroll
    for (int i = 0; i < 4; ++i) {                 // coalesced b128 stores
        const int c = tid + 256 * i;              // 1024 chunks of 8 halfs
        const int r = c >> 3, off = (c & 7) << 3;
        *(h8*)(out + (rowbase + r) * 768 + h * 64 + off) = *(const h8*)(Os + r * 72 + off);
    }
}

// ---------------------------------------------------------------------------
// Host-side orchestration.
// Workspace layout (bytes):
//   [0, 50.3M)          h16 / h2      f16 [32768,768] (reused for LN2 out)
//   [50.3M, 246.9M)     qkv f16 [32768,2304]; later mlp1 f16 [32000,3072]
//   [201.3M, 251.7M)    attn_out f16 [32768,768] (dead before mlp1 reaches it)
//   [251.7M, 350.0M)    xres fp32 [32000,768]
//   [350.0M, 364.1M)    f16-transposed weights (qkv, proj, fc1, fc2)
// ---------------------------------------------------------------------------
extern "C" void kernel_launch(void* const* d_in, const int* in_sizes, int n_in,
                              void* d_out, int out_size, void* d_ws, size_t ws_size,
                              hipStream_t stream)
{
    const float* x       = (const float*)d_in[0];
    const float* n1w     = (const float*)d_in[1];
    const float* n1b     = (const float*)d_in[2];
    const float* qkv_w   = (const float*)d_in[3];
    const float* qkv_b   = (const float*)d_in[4];
    const float* proj_w  = (const float*)d_in[5];
    const float* proj_b  = (const float*)d_in[6];
    const float* relb    = (const float*)d_in[7];
    const float* ls1     = (const float*)d_in[8];
    const float* n2w     = (const float*)d_in[9];
    const float* n2b     = (const float*)d_in[10];
    const float* fc1_w   = (const float*)d_in[11];
    const float* fc1_b   = (const float*)d_in[12];
    const float* fc2_w   = (const float*)d_in[13];
    const float* fc2_b   = (const float*)d_in[14];
    const float* ls2     = (const float*)d_in[15];

    char* ws = (char*)d_ws;
    _Float16* h16_   = (_Float16*)ws;                        // also h2
    _Float16* qkv16  = (_Float16*)(ws + 50331648ull);        // also mlp1
    _Float16* attn16 = (_Float16*)(ws + 201326592ull);
    float*    xres   = (float*)   (ws + 251658240ull);
    _Float16* wq16   = (_Float16*)(ws + 349962240ull);       // [2304, 768]
    _Float16* wp16   = (_Float16*)(ws + 353501184ull);       // [768, 768]
    _Float16* w1_16  = (_Float16*)(ws + 354680832ull);       // [3072, 768]
    _Float16* w2_16  = (_Float16*)(ws + 359399424ull);       // [768, 3072]

    // 0) Weights -> f16, transposed [N][K]
    transpose_w16<<<6912, 256, 0, stream>>>(qkv_w,  wq16,  768, 2304);
    transpose_w16<<<2304, 256, 0, stream>>>(proj_w, wp16,  768, 768);
    transpose_w16<<<9216, 256, 0, stream>>>(fc1_w,  w1_16, 768, 3072);
    transpose_w16<<<9216, 256, 0, stream>>>(fc2_w,  w2_16, 3072, 768);

    // 1) LN1 fused with pad + roll(-64) -> f16 rows [16*2048, 768]
    layernorm_f16<true><<<4096, 256, 0, stream>>>(x, n1w, n1b, h16_, 32768);

    // 2) QKV GEMM: [32768,768] x [768,2304] -> f16
    gemm_wmma<0><<<dim3(256, 18), 256, 0, stream>>>(
        h16_, wq16, qkv_b, qkv16, nullptr, nullptr, nullptr, 768, 2304);

    // 3) Windowed attention: 256 windows x 12 heads
    window_attn<<<3072, 256, 0, stream>>>(qkv16, relb, attn16);

    // 4) Proj GEMM + un-roll + crop + residual: xres = x + a*ls1
    gemm_wmma<1><<<dim3(256, 6), 256, 0, stream>>>(
        attn16, wp16, proj_b, nullptr, xres, x, ls1, 768, 768);

    // 5) LN2 -> f16 [32000,768]
    layernorm_f16<false><<<4000, 256, 0, stream>>>(xres, n2w, n2b, h16_, 32000);

    // 6) FC1 GEMM + exact GELU -> f16 [32000,3072]
    gemm_wmma<2><<<dim3(250, 24), 256, 0, stream>>>(
        h16_, w1_16, fc1_b, qkv16, nullptr, nullptr, nullptr, 768, 3072);

    // 7) FC2 GEMM + residual: out = xres + m*ls2
    gemm_wmma<3><<<dim3(250, 6), 256, 0, stream>>>(
        qkv16, w2_16, fc2_b, nullptr, (float*)d_out, xres, ls2, 3072, 768);
}
